// TPFN_1992864825688
// MI455X (gfx1250) — compile-verified, matrix operand order checked by reading, not run on hardware
//
#include <hip/hip_runtime.h>

// ---------------------------------------------------------------------------
// Problem constants (match reference)
// ---------------------------------------------------------------------------
#define BB      256
#define TT      128
#define LENZ    512     // A_HID + V_HID + T_HID
#define RO      2048    // RANK * OUT_DIM
#define KSEL    64      // T / STRIDE
#define T_IN    768
#define A_IN    74
#define V_IN    35
#define KT_PAD  768     // multiples of 32 for WMMA K loop
#define KA_PAD  96
#define KV_PAD  64
#define EXT_DIM 1025

typedef __attribute__((ext_vector_type(16))) _Float16 v16h;
typedef __attribute__((ext_vector_type(8)))  _Float16 v8h;
typedef __attribute__((ext_vector_type(8)))  float    v8f;

#define DEVINL __device__ __forceinline__

// ---------------------------------------------------------------------------
// WMMA fragment helpers.
// CDNA5 16-bit A/B fragment layout (ISA 7.12.2): for lane L, the 16 halves are
// two contiguous 8-half chunks of row (L&15):  K = [k+hi8, k+hi8+8) and
// [k+16+hi8, k+24+hi8) with hi8 = (L>>4)*8.  Because every GEMM here is
// X(row-major, K contiguous) x W(row-major, K contiguous)^T, A and B frags use
// the *same* loader.
// ---------------------------------------------------------------------------
DEVINL v16h ldfrag(const _Float16* __restrict__ rowp, int k, int hi8) {
    const v8h* p0 = reinterpret_cast<const v8h*>(rowp + k + hi8);
    const v8h* p1 = reinterpret_cast<const v8h*>(rowp + k + 16 + hi8);
    v8h lo = *p0, hi = *p1;
    v16h r;
#pragma unroll
    for (int i = 0; i < 8; ++i) { r[i] = lo[i]; r[i + 8] = hi[i]; }
    return r;
}

DEVINL v8f wmma_f16(v16h a, v16h b, v8f c) {
    // D = A(16x32 f16) * B(32x16 f16) + C(16x16 f32)
    return __builtin_amdgcn_wmma_f32_16x16x32_f16(
        false, a, false, b, (short)0, c, false, false);
}

DEVINL float sigf(float x) { return 1.0f / (1.0f + __expf(-x)); }

// ---------------------------------------------------------------------------
// K0: fp32 -> f16 pack with column padding / slicing (for W2 split).
// dst is rows x dst_cols (dense); src read at src[r*src_ld + src_col0 + c].
// ---------------------------------------------------------------------------
__global__ void pack_f16_kernel(const float* __restrict__ src,
                                _Float16* __restrict__ dst,
                                int rows, int src_cols, int dst_cols,
                                int src_ld, int src_col0) {
    size_t idx   = (size_t)blockIdx.x * blockDim.x + threadIdx.x;
    size_t total = (size_t)rows * dst_cols;
    if (idx >= total) return;
    int r = (int)(idx / dst_cols);
    int c = (int)(idx % dst_cols);
    float v = (c < src_cols) ? src[(size_t)r * src_ld + src_col0 + c] : 0.0f;
    dst[idx] = (_Float16)v;
}

// ---------------------------------------------------------------------------
// K1: out(MxN,f32) = A(MxK,f16) * B(NxK,f16)^T + bias(N)
// Block: 256 threads = 8 waves. Block tile = 16(M) x 128(N); each wave owns a
// 16x16 accumulator and streams K in steps of 32.
// ---------------------------------------------------------------------------
__global__ void wmma_gemm_bias_kernel(const _Float16* __restrict__ A,
                                      const _Float16* __restrict__ Bm,
                                      const float* __restrict__ bias,
                                      float* __restrict__ out,
                                      int K, int N) {
    const int lane = threadIdx.x & 31;
    const int wave = threadIdx.x >> 5;
    const int lo   = lane & 15;
    const int hi8  = (lane >> 4) * 8;
    const int row0 = blockIdx.x * 16;
    const int col0 = blockIdx.y * 128 + wave * 16;

    const _Float16* arow = A  + (size_t)(row0 + lo) * K;
    const _Float16* brow = Bm + (size_t)(col0 + lo) * K;

    const float bv = bias ? bias[col0 + lo] : 0.0f;
    v8f acc = {bv, bv, bv, bv, bv, bv, bv, bv};   // C layout: fixed N per lane

    for (int k = 0; k < K; k += 32) {
        __builtin_prefetch(brow + k + 128, 0, 1);   // global_prefetch_b8
        v16h af = ldfrag(arow, k, hi8);
        v16h bf = ldfrag(brow, k, hi8);
        acc = wmma_f16(af, bf, acc);
    }
    // D layout: VGPR j holds M = j + (lane>=16 ? 8 : 0), N = lane&15
    const int mtop = row0 + hi8;   // hi8 is exactly 0 or 8
#pragma unroll
    for (int j = 0; j < 8; ++j)
        out[(size_t)(mtop + j) * N + col0 + lo] = acc[j];
}

// ---------------------------------------------------------------------------
// K2: LSTM recurrence for one modality. One block = 16 batch rows, all H.
// xp: (B,T,4H) f32 precomputed x-projection (+bias). Whh: (4H,H) f16.
// z: (B,T,512) f16 concat hidden output, written at column zoff.
// Dynamic LDS: hbuf 16*H f16 | cbuf 16*H f32 | gbuf 16*4H f32.
// ---------------------------------------------------------------------------
template <int H>
__global__ void lstm_kernel(const float* __restrict__ xp,
                            const _Float16* __restrict__ Whh,
                            _Float16* __restrict__ z,
                            int zoff, int Tn) {
    constexpr int G   = 4 * H;
    constexpr int TPW = G / (8 * 16);   // 16x16 tiles per wave (8 or 4)

    extern __shared__ char smem[];
    _Float16* hbuf = (_Float16*)smem;                                   // 16*H
    float*    cbuf = (float*)(smem + (size_t)16 * H * sizeof(_Float16));
    float*    gbuf = (float*)(smem + (size_t)16 * H * (sizeof(_Float16) + sizeof(float)));

    const int tid  = threadIdx.x;
    const int lane = tid & 31;
    const int wave = tid >> 5;
    const int lo   = lane & 15;
    const int hi8  = (lane >> 4) * 8;
    const int b0   = blockIdx.x * 16;
    const int colw = wave * TPW * 16;

    for (int e = tid; e < 16 * H; e += 256) { hbuf[e] = (_Float16)0.0f; cbuf[e] = 0.0f; }
    __syncthreads();

    for (int t = 0; t < Tn; ++t) {
        // ---- phase 1: g = xp[:,t,:] + h * Whh^T  (WMMA over K=H) ----
        v8f acc[TPW];
#pragma unroll
        for (int nt = 0; nt < TPW; ++nt) {
            const int col = colw + nt * 16 + lo;
#pragma unroll
            for (int j = 0; j < 8; ++j)
                acc[nt][j] = xp[((size_t)(b0 + hi8 + j) * Tn + t) * G + col];
        }
        const _Float16* hrow = &hbuf[lo * H];
        for (int k = 0; k < H; k += 32) {
            v16h af = ldfrag(hrow, k, hi8);      // ds loads from LDS
#pragma unroll
            for (int nt = 0; nt < TPW; ++nt) {
                const _Float16* wrow = Whh + (size_t)(colw + nt * 16 + lo) * H;
                v16h bf = ldfrag(wrow, k, hi8);
                acc[nt] = wmma_f16(af, bf, acc[nt]);
            }
        }
#pragma unroll
        for (int nt = 0; nt < TPW; ++nt) {
            const int col = colw + nt * 16 + lo;
#pragma unroll
            for (int j = 0; j < 8; ++j)
                gbuf[(hi8 + j) * G + col] = acc[nt][j];
        }
        __syncthreads();

        // ---- phase 2: gate nonlinearities, state update, emit h ----
        for (int e = tid; e < 16 * H; e += 256) {
            const int r = e / H, c = e % H;
            const float gi = gbuf[r * G + c];
            const float gf = gbuf[r * G + c + H];
            const float gg = gbuf[r * G + c + 2 * H];
            const float go = gbuf[r * G + c + 3 * H];
            const float cn = sigf(gf) * cbuf[e] + sigf(gi) * tanhf(gg);
            cbuf[e] = cn;
            const float hn = sigf(go) * tanhf(cn);
            hbuf[e] = (_Float16)hn;
            z[((size_t)(b0 + r) * Tn + t) * LENZ + zoff + c] = (_Float16)hn;
        }
        __syncthreads();
    }
}

// ---------------------------------------------------------------------------
// K3: fused low-rank tensor fusion.
// f1[b,k,:] = b1 + z[b,2k]   * W1^T
// f2[b,k,:] = b2 + W2[:,0] + z[b,2k] * W2a^T + z[b,2k+1] * W2b^T
// zf[b,:]   = mean_k (f1 .* f2)
// Grid: (B, RO/128). 8 waves, each owns one 16-col tile across 4 k-tiles.
// ---------------------------------------------------------------------------
__global__ void fusion_kernel(const _Float16* __restrict__ z,
                              const _Float16* __restrict__ W1h,
                              const _Float16* __restrict__ W2a,
                              const _Float16* __restrict__ W2b,
                              const float* __restrict__ b1,
                              const float* __restrict__ b2,
                              const float* __restrict__ W2full,
                              float* __restrict__ zf, int Tn) {
    const int lane = threadIdx.x & 31;
    const int wave = threadIdx.x >> 5;
    const int lo   = lane & 15;
    const int hi8  = (lane >> 4) * 8;
    const int b    = blockIdx.x;
    const int n    = blockIdx.y * 128 + wave * 16 + lo;

    const float bias1 = b1[n];
    const float bias2 = b2[n] + W2full[(size_t)n * EXT_DIM];   // ones column

    const _Float16* w1r  = W1h + (size_t)n * LENZ;
    const _Float16* w2ar = W2a + (size_t)n * LENZ;
    const _Float16* w2br = W2b + (size_t)n * LENZ;

    v8f zfacc = {0, 0, 0, 0, 0, 0, 0, 0};
#pragma unroll
    for (int mt = 0; mt < KSEL / 16; ++mt) {
        const int kk = mt * 16 + lo;               // selected k index (row M)
        const _Float16* za = z + ((size_t)b * Tn + 2 * kk)     * LENZ;
        const _Float16* zb = z + ((size_t)b * Tn + 2 * kk + 1) * LENZ;
        v8f a1 = {bias1, bias1, bias1, bias1, bias1, bias1, bias1, bias1};
        v8f a2 = {bias2, bias2, bias2, bias2, bias2, bias2, bias2, bias2};
        for (int k = 0; k < LENZ; k += 32) {
            __builtin_prefetch(w2br + k + 128, 0, 1);
            v16h fe = ldfrag(za, k, hi8);          // z[b, 2k]
            a1 = wmma_f16(fe, ldfrag(w1r,  k, hi8), a1);
            a2 = wmma_f16(fe, ldfrag(w2ar, k, hi8), a2);
            v16h fo = ldfrag(zb, k, hi8);          // z[b, 2k+1]
            a2 = wmma_f16(fo, ldfrag(w2br, k, hi8), a2);
        }
#pragma unroll
        for (int j = 0; j < 8; ++j) zfacc[j] += a1[j] * a2[j];
    }
    // reduce over the 16 k-rows of each tile: 8 in-register + cross-half shuffle
    float s = 0.0f;
#pragma unroll
    for (int j = 0; j < 8; ++j) s += zfacc[j];
    s += __shfl_xor(s, 16, 32);
    if (lane < 16)
        zf[(size_t)b * RO + n] = s * (1.0f / (float)KSEL);
}

// ---------------------------------------------------------------------------
// K4: output[b,o] = sum_r fusion_w[r] * zf[b, r*16 + o]
// ---------------------------------------------------------------------------
__global__ void out_kernel(const float* __restrict__ zf,
                           const float* __restrict__ fw,
                           float* __restrict__ out) {
    const int idx = blockIdx.x * blockDim.x + threadIdx.x;   // B*16
    const int b = idx >> 4, o = idx & 15;
    float s = 0.0f;
    for (int r = 0; r < 128; ++r)
        s += fw[r] * zf[(size_t)b * RO + r * 16 + o];
    out[idx] = s;
}

// ---------------------------------------------------------------------------
// Host side
// ---------------------------------------------------------------------------
static inline size_t al256(size_t x) { return (x + 255) & ~(size_t)255; }

extern "C" void kernel_launch(void* const* d_in, const int* in_sizes, int n_in,
                              void* d_out, int out_size, void* d_ws, size_t ws_size,
                              hipStream_t stream) {
    (void)in_sizes; (void)n_in; (void)out_size; (void)ws_size;

    const float* text_x  = (const float*)d_in[0];
    const float* audio_x = (const float*)d_in[1];
    const float* video_x = (const float*)d_in[2];
    const float* W_ih_t  = (const float*)d_in[3];
    const float* W_hh_t  = (const float*)d_in[4];
    const float* b_t     = (const float*)d_in[5];
    const float* W_ih_a  = (const float*)d_in[6];
    const float* W_hh_a  = (const float*)d_in[7];
    const float* b_a     = (const float*)d_in[8];
    const float* W_ih_v  = (const float*)d_in[9];
    const float* W_hh_v  = (const float*)d_in[10];
    const float* b_v     = (const float*)d_in[11];
    const float* W1      = (const float*)d_in[12];
    const float* b1      = (const float*)d_in[13];
    const float* W2      = (const float*)d_in[14];
    const float* b2      = (const float*)d_in[15];
    const float* fw      = (const float*)d_in[16];
    float* out = (float*)d_out;

    // ---- workspace layout ----
    char* wsb = (char*)d_ws;
    size_t off = 0;
    auto take = [&](size_t bytes) { char* p = wsb + off; off += al256(bytes); return p; };
    const size_t MR = (size_t)BB * TT;   // 32768 rows

    _Float16* xt16  = (_Float16*)take(MR * KT_PAD * 2);
    _Float16* xa16  = (_Float16*)take(MR * KA_PAD * 2);
    _Float16* xv16  = (_Float16*)take(MR * KV_PAD * 2);
    _Float16* wiht  = (_Float16*)take((size_t)1024 * KT_PAD * 2);
    _Float16* wiha  = (_Float16*)take((size_t)512  * KA_PAD * 2);
    _Float16* wihv  = (_Float16*)take((size_t)512  * KV_PAD * 2);
    _Float16* whht  = (_Float16*)take((size_t)1024 * 256 * 2);
    _Float16* whha  = (_Float16*)take((size_t)512  * 128 * 2);
    _Float16* whhv  = (_Float16*)take((size_t)512  * 128 * 2);
    _Float16* w1h   = (_Float16*)take((size_t)RO * LENZ * 2);
    _Float16* w2a   = (_Float16*)take((size_t)RO * LENZ * 2);
    _Float16* w2b   = (_Float16*)take((size_t)RO * LENZ * 2);
    float*    xp_t  = (float*)take(MR * 1024 * 4);
    float*    xp_a  = (float*)take(MR * 512 * 4);
    float*    xp_v  = (float*)take(MR * 512 * 4);
    _Float16* zbuf  = (_Float16*)take((size_t)BB * TT * LENZ * 2);
    float*    zf    = (float*)take((size_t)BB * RO * 4);

    auto pack = [&](const float* src, _Float16* dst, int rows, int sc, int dc,
                    int sld, int sc0) {
        size_t total = (size_t)rows * dc;
        int grid = (int)((total + 255) / 256);
        pack_f16_kernel<<<grid, 256, 0, stream>>>(src, dst, rows, sc, dc, sld, sc0);
    };

    // ---- 0: pack / pad everything to f16 ----
    pack(text_x,  xt16, (int)MR, T_IN, KT_PAD, T_IN, 0);
    pack(audio_x, xa16, (int)MR, A_IN, KA_PAD, A_IN, 0);
    pack(video_x, xv16, (int)MR, V_IN, KV_PAD, V_IN, 0);
    pack(W_ih_t, wiht, 1024, T_IN, KT_PAD, T_IN, 0);
    pack(W_ih_a, wiha, 512,  A_IN, KA_PAD, A_IN, 0);
    pack(W_ih_v, wihv, 512,  V_IN, KV_PAD, V_IN, 0);
    pack(W_hh_t, whht, 1024, 256, 256, 256, 0);
    pack(W_hh_a, whha, 512,  128, 128, 128, 0);
    pack(W_hh_v, whhv, 512,  128, 128, 128, 0);
    pack(W1, w1h, RO, LENZ, LENZ, LENZ, 0);
    pack(W2, w2a, RO, LENZ, LENZ, EXT_DIM, 1);          // W2[:,1:513]
    pack(W2, w2b, RO, LENZ, LENZ, EXT_DIM, 1 + LENZ);   // W2[:,513:1025]

    // ---- 1: xp = x * W_ih^T + b  (WMMA GEMMs) ----
    wmma_gemm_bias_kernel<<<dim3((unsigned)(MR / 16), 1024 / 128), 256, 0, stream>>>(
        xt16, wiht, b_t, xp_t, KT_PAD, 1024);
    wmma_gemm_bias_kernel<<<dim3((unsigned)(MR / 16), 512 / 128), 256, 0, stream>>>(
        xa16, wiha, b_a, xp_a, KA_PAD, 512);
    wmma_gemm_bias_kernel<<<dim3((unsigned)(MR / 16), 512 / 128), 256, 0, stream>>>(
        xv16, wihv, b_v, xp_v, KV_PAD, 512);

    // ---- 2: recurrences (write into z at concat offsets: audio|video|text) ----
    lstm_kernel<128><<<BB / 16, 256, 16 * 128 * (2 + 4) + 16 * 512  * 4, stream>>>(
        xp_a, whha, zbuf, 0, TT);
    lstm_kernel<128><<<BB / 16, 256, 16 * 128 * (2 + 4) + 16 * 512  * 4, stream>>>(
        xp_v, whhv, zbuf, 128, TT);
    lstm_kernel<256><<<BB / 16, 256, 16 * 256 * (2 + 4) + 16 * 1024 * 4, stream>>>(
        xp_t, whht, zbuf, 256, TT);

    // ---- 3: fused f1.*f2 mean ----
    fusion_kernel<<<dim3(BB, RO / 128), 256, 0, stream>>>(
        zbuf, w1h, w2a, w2b, b1, b2, W2, zf, TT);

    // ---- 4: final contraction ----
    out_kernel<<<(BB * 16) / 256, 256, 0, stream>>>(zf, fw, out);
}
